// GCNModel_86732569575634
// MI455X (gfx1250) — 16-bit and 8-bit matrix operand order repaired and checked by reading, compiled, not run on hardware
//
#include <hip/hip_runtime.h>

typedef __attribute__((ext_vector_type(16))) int    v16i;
typedef __attribute__((ext_vector_type(8)))  float  v8f;
typedef __attribute__((ext_vector_type(16))) __bf16 v16bf;

union U16 { unsigned u[8]; v16bf v; };

#define NNODE 3136
#define NPG   784
#define NG    4
#define NEDGE 25088
#define LDA   896
#define KT_ADJ 7
#define MTILES 49
#define THRESH2 0.09f
#define FP8_ONE 0x38

__global__ void fill_f32(float* p, float v, int n) {
    int i = blockIdx.x * blockDim.x + threadIdx.x;
    if (i < n) p[i] = v;
}
__global__ void fill_u32(unsigned* p, unsigned v, int n) {
    int i = blockIdx.x * blockDim.x + threadIdx.x;
    if (i < n) p[i] = v;
}
__global__ void fill_i32(int* p, int v, int n) {
    int i = blockIdx.x * blockDim.x + threadIdx.x;
    if (i < n) p[i] = v;
}

__global__ void conv_weights(const float* w0, const float* w1, const float* w2, __bf16* wt) {
    int i = blockIdx.x * blockDim.x + threadIdx.x;
    if (i >= 3 * 4096) return;
    int which = i >> 12, r = i & 4095;
    int k = r >> 6, n = r & 63;
    const float* w = (which == 0) ? w0 : (which == 1) ? w1 : w2;
    wt[which * 4096 + n * 64 + k] = (__bf16)w[k * 64 + n];
}

__global__ void xw_k(const float* x, const float* W1, float* xw) {
    int i = blockIdx.x * blockDim.x + threadIdx.x;
    if (i >= NNODE * 64) return;
    int v = i >> 6, o = i & 63;
    float s = 0.f;
    #pragma unroll
    for (int k = 0; k < 3; ++k) s += x[v * 3 + k] * W1[k * 64 + o];
    xw[i] = s;
}

__global__ void deg_k(const int* dst, float* deg) {
    int e = blockIdx.x * blockDim.x + threadIdx.x;
    if (e < NEDGE) atomicAdd(&deg[dst[e]], 1.0f);
}
__global__ void dinv_k(const float* deg, float* dinv) {
    int v = blockIdx.x * blockDim.x + threadIdx.x;
    if (v < NNODE) dinv[v] = rsqrtf(deg[v] + 1.0f);
}
__global__ void agg_k(const int* src, const int* dst, const float* dinv,
                      const float* hin, float* agg) {
    int i = blockIdx.x * blockDim.x + threadIdx.x;
    if (i >= NEDGE * 64) return;
    int e = i >> 6, f = i & 63;
    int s = src[e], d = dst[e];
    float nrm = dinv[s] * dinv[d];
    atomicAdd(&agg[(size_t)d * 64 + f], hin[(size_t)s * 64 + f] * nrm);
}
__global__ void gcn_fin(const float* agg, const float* hw, const float* dinv,
                        const float* bias, float* hout, __bf16* hb, float* sq) {
    int i = blockIdx.x * blockDim.x + threadIdx.x;
    if (i >= NNODE * 64) return;
    int v = i >> 6, f = i & 63;
    float di = dinv[v];
    float val = agg[i] + hw[i] * di * di + bias[f];
    val = fmaxf(val, 0.f);
    hout[i] = val;
    if (hb) hb[i] = (__bf16)val;
    if (sq) atomicAdd(&sq[v], val * val);
}

__global__ void build_adj(const int* src, const int* dst, unsigned char* adj, unsigned char* adjT) {
    int e = blockIdx.x * blockDim.x + threadIdx.x;
    if (e >= NEDGE) return;
    int s = src[e], d = dst[e];
    int g = s / NPG;
    int ls = s - g * NPG, ld = d - g * NPG;
    size_t base = (size_t)g * NPG * LDA;
    adj [base + (size_t)ls * LDA + ld] = FP8_ONE;
    adjT[base + (size_t)ld * LDA + ls] = FP8_ONE;
}

__global__ __launch_bounds__(256)
void pool_tiles(const unsigned char* adj, const unsigned char* adjT,
                const __bf16* hb, const float* sq, int* cand) {
    int wave = blockIdx.x * (blockDim.x >> 5) + (threadIdx.x >> 5);
    if (wave >= NG * MTILES * MTILES) return;
    int g  = wave / (MTILES * MTILES);
    int mt = (wave / MTILES) % MTILES;
    int nt = wave % MTILES;
    int l = threadIdx.x & 31, half = l >> 4, lo = l & 15;

    const unsigned char* A = adj  + (size_t)g * NPG * LDA + (size_t)(mt * 16 + lo) * LDA;
    const unsigned char* B = adjT + (size_t)g * NPG * LDA + (size_t)(nt * 16 + lo) * LDA;

    auto loadA = [&](int K0, v16i& a) {
        #pragma unroll
        for (int v = 0; v < 16; ++v) {
            int w = v & 7;
            int ka = K0 + (v >> 3) * 64 + (w >> 1) * 16 + half * 8 + (w & 1) * 4;
            a[v] = *(const int*)(A + ka);
        }
    };
    auto loadB = [&](int K0, v16i& b) {
        #pragma unroll
        for (int v = 0; v < 16; ++v) {
            int kb = K0 + (v >> 3) * 64 + ((v & 7) >> 1) * 16 + half * 8 + (v & 1) * 4;
            b[v] = *(const int*)(B + kb);
        }
    };

    v8f hop = {};
    v16i a0, a1, b0, b1;
    loadA(0, a0);
    loadB(0, b0);
    #pragma unroll
    for (int kt = 0; kt < KT_ADJ; ++kt) {
        if (kt + 1 < KT_ADJ) {
            if (kt & 1) { loadA((kt + 1) * 128, a0); loadB((kt + 1) * 128, b0); }
            else        { loadA((kt + 1) * 128, a1); loadB((kt + 1) * 128, b1); }
        }
        const v16i& ca = (kt & 1) ? a1 : a0;
        const v16i& cb = (kt & 1) ? b1 : b0;
        hop = __builtin_amdgcn_wmma_f32_16x16x128_fp8_fp8(ca, cb, (short)0, hop, false, false);
    }

    v8f d = {};
    const __bf16* Hi = hb + (size_t)(g * NPG + mt * 16 + lo) * 64;
    const __bf16* Hj = hb + (size_t)(g * NPG + nt * 16 + lo) * 64;
    #pragma unroll
    for (int kt = 0; kt < 2; ++kt) {
        int K0 = kt * 32;
        U16 au, bu;
        #pragma unroll
        for (int v = 0; v < 8; ++v) {
            int ka = K0 + (v >> 2) * 16 + half * 8 + (v & 3) * 2;
            int kb = K0 + (v >> 2) * 16 + half * 8 + (v & 3) * 2;
            au.u[v] = *(const unsigned*)(Hi + ka);
            bu.u[v] = *(const unsigned*)(Hj + kb);
        }
        d = __builtin_amdgcn_wmma_f32_16x16x32_bf16(false, au.v, false, bu.v,
                                                    (short)0, d, false, false);
    }

    int jg = g * NPG + nt * 16 + lo;
    float sqj = sq[jg];
    int best = 0x7FFFFFFF;
    #pragma unroll
    for (int r = 0; r < 8; ++r) {
        int ig = g * NPG + mt * 16 + r + half * 8;
        float d2 = fmaxf(sq[ig] + sqj - 2.0f * d[r], 0.0f);
        if (hop[r] > 0.5f && d2 < THRESH2 && ig != jg) best = (ig < best) ? ig : best;
    }
    int other = __shfl_xor(best, 16, 32);
    best = (other < best) ? other : best;
    if (half == 0) atomicMin(&cand[jg], best);
}

__global__ void rep_k(const int* cand, int* rep) {
    int v = blockIdx.x * blockDim.x + threadIdx.x;
    if (v < NNODE) rep[v] = (cand[v] < v) ? cand[v] : v;
}
__global__ void pool_accum(const int* rep, const float* h, float* psum, float* pcount) {
    int i = blockIdx.x * blockDim.x + threadIdx.x;
    if (i >= NNODE * 64) return;
    int v = i >> 6, f = i & 63;
    int r = rep[v];
    atomicAdd(&psum[(size_t)r * 64 + f], h[i]);
    if (f == 0) atomicAdd(&pcount[r], 1.0f);
}
__global__ void pool_final(const float* psum, const float* pcount,
                           float* newx, __bf16* newxb, float* valid) {
    int i = blockIdx.x * blockDim.x + threadIdx.x;
    if (i >= NNODE * 64) return;
    int v = i >> 6, f = i & 63;
    float cnt = pcount[v];
    float nx = psum[i] / fmaxf(cnt, 1.0f);
    newx[i] = nx;
    newxb[i] = (__bf16)nx;
    if (f == 0) valid[v] = (cnt > 0.f) ? 1.f : 0.f;
}

__global__ __launch_bounds__(256)
void gemm64(const __bf16* A, const __bf16* WT, const float* bias,
            float* outF, __bf16* outB, int relu) {
    int wave = blockIdx.x * (blockDim.x >> 5) + (threadIdx.x >> 5);
    if (wave >= (NNODE / 16) * 4) return;
    int mt = wave >> 2, nt = wave & 3;
    int l = threadIdx.x & 31, half = l >> 4, lo = l & 15;

    const __bf16* Ar = A  + (size_t)(mt * 16 + lo) * 64;
    const __bf16* Bc = WT + (size_t)(nt * 16 + lo) * 64;

    v8f acc = {};
    #pragma unroll
    for (int kt = 0; kt < 2; ++kt) {
        int K0 = kt * 32;
        U16 au, bu;
        #pragma unroll
        for (int v = 0; v < 8; ++v) {
            int ka = K0 + (v >> 2) * 16 + half * 8 + (v & 3) * 2;
            int kb = K0 + (v >> 2) * 16 + half * 8 + (v & 3) * 2;
            au.u[v] = *(const unsigned*)(Ar + ka);
            bu.u[v] = *(const unsigned*)(Bc + kb);
        }
        acc = __builtin_amdgcn_wmma_f32_16x16x32_bf16(false, au.v, false, bu.v,
                                                      (short)0, acc, false, false);
    }

    int col = nt * 16 + lo;
    float b = bias ? bias[col] : 0.f;
    #pragma unroll
    for (int r = 0; r < 8; ++r) {
        int row = mt * 16 + r + half * 8;
        float v = acc[r] + b;
        if (relu) v = fmaxf(v, 0.f);
        outF[(size_t)row * 64 + col] = v;
        if (outB) outB[(size_t)row * 64 + col] = (__bf16)v;
    }
}

__global__ void deg2_k(const int* rep, const int* dst, float* deg) {
    int e = blockIdx.x * blockDim.x + threadIdx.x;
    if (e < NEDGE) atomicAdd(&deg[rep[dst[e]]], 1.0f);
}
__global__ void agg2_k(const int* rep, const int* src, const int* dst, const float* dinv,
                       const float* hin, float* agg) {
    int i = blockIdx.x * blockDim.x + threadIdx.x;
    if (i >= NEDGE * 64) return;
    int e = i >> 6, f = i & 63;
    int rs = rep[src[e]], rd = rep[dst[e]];
    float nrm = dinv[rs] * dinv[rd];
    atomicAdd(&agg[(size_t)rd * 64 + f], hin[(size_t)rs * 64 + f] * nrm);
}

__global__ void pool_cls(const float* h4, const float* valid,
                         const float* clsW, const float* clsb, float* out) {
    int g = blockIdx.x, f = threadIdx.x;
    __shared__ float mean[64];
    __shared__ float logits[10];
    float num = 0.f, den = 0.f;
    for (int v = 0; v < NPG; ++v) {
        float w = valid[g * NPG + v];
        num += h4[(size_t)(g * NPG + v) * 64 + f] * w;
        den += w;
    }
    mean[f] = num / fmaxf(den, 1.0f);
    __syncthreads();
    if (f < 10) {
        float s = clsb[f];
        for (int k = 0; k < 64; ++k) s += mean[k] * clsW[k * 10 + f];
        logits[f] = s;
    }
    __syncthreads();
    if (f == 0) {
        float mx = -1e30f;
        for (int c = 0; c < 10; ++c) mx = fmaxf(mx, logits[c]);
        float se = 0.f;
        for (int c = 0; c < 10; ++c) se += __expf(logits[c] - mx);
        float lse = mx + __logf(se);
        for (int c = 0; c < 10; ++c) out[g * 10 + c] = logits[c] - lse;
    }
}

extern "C" void kernel_launch(void* const* d_in, const int* in_sizes, int n_in,
                              void* d_out, int out_size, void* d_ws, size_t ws_size,
                              hipStream_t stream) {
    const float* x     = (const float*)d_in[0];
    const int*   ei    = (const int*)  d_in[1];
    const float* W1    = (const float*)d_in[3];
    const float* b1    = (const float*)d_in[4];
    const float* lin1W = (const float*)d_in[5];
    const float* lin1b = (const float*)d_in[6];
    const float* W2    = (const float*)d_in[7];
    const float* b2    = (const float*)d_in[8];
    const float* lin2W = (const float*)d_in[9];
    const float* lin2b = (const float*)d_in[10];
    const float* clsW  = (const float*)d_in[11];
    const float* clsb  = (const float*)d_in[12];
    const int* src = ei;
    const int* dst = ei + NEDGE;
    float* out = (float*)d_out;
    (void)in_sizes; (void)n_in; (void)out_size; (void)ws_size;

    char* ws = (char*)d_ws;
    size_t off = 0;
    auto take = [&](size_t bytes) -> char* {
        char* p = ws + off;
        off = (off + bytes + 255) & ~(size_t)255;
        return p;
    };
    const size_t ADJB = (size_t)NG * NPG * LDA;
    const size_t NF   = (size_t)NNODE * 64;
    unsigned char* adj  = (unsigned char*)take(ADJB);
    unsigned char* adjT = (unsigned char*)take(ADJB);
    __bf16* hb1   = (__bf16*)take(NF * 2);
    __bf16* hb2   = (__bf16*)take(NF * 2);
    __bf16* hb3   = (__bf16*)take(NF * 2);
    __bf16* hb4   = (__bf16*)take(NF * 2);
    __bf16* wt    = (__bf16*)take(3 * 4096 * 2);
    float* xw     = (float*)take(NF * 4);
    float* agg    = (float*)take(NF * 4);
    float* h      = (float*)take(NF * 4);
    float* psum   = (float*)take(NF * 4);
    float* newx   = (float*)take(NF * 4);
    float* h2     = (float*)take(NF * 4);
    float* hw2    = (float*)take(NF * 4);
    float* agg2   = (float*)take(NF * 4);
    float* h3     = (float*)take(NF * 4);
    float* h4     = (float*)take(NF * 4);
    float* deg    = (float*)take(NNODE * 4);
    float* dinv   = (float*)take(NNODE * 4);
    float* sq     = (float*)take(NNODE * 4);
    float* pcount = (float*)take(NNODE * 4);
    float* deg2   = (float*)take(NNODE * 4);
    float* dinv2  = (float*)take(NNODE * 4);
    float* valid  = (float*)take(NNODE * 4);
    int*   cand   = (int*)  take(NNODE * 4);
    int*   rep    = (int*)  take(NNODE * 4);

    const int T = 256;
    auto blk = [](long n, int t) { return (int)((n + t - 1) / t); };

    fill_u32<<<blk(ADJB / 4, T), T, 0, stream>>>((unsigned*)adj, 0u, (int)(ADJB / 4));
    fill_u32<<<blk(ADJB / 4, T), T, 0, stream>>>((unsigned*)adjT, 0u, (int)(ADJB / 4));
    fill_f32<<<blk(NF, T), T, 0, stream>>>(agg, 0.f, (int)NF);
    fill_f32<<<blk(NF, T), T, 0, stream>>>(psum, 0.f, (int)NF);
    fill_f32<<<blk(NF, T), T, 0, stream>>>(agg2, 0.f, (int)NF);
    fill_f32<<<blk(NNODE, T), T, 0, stream>>>(deg, 0.f, NNODE);
    fill_f32<<<blk(NNODE, T), T, 0, stream>>>(sq, 0.f, NNODE);
    fill_f32<<<blk(NNODE, T), T, 0, stream>>>(pcount, 0.f, NNODE);
    fill_f32<<<blk(NNODE, T), T, 0, stream>>>(deg2, 0.f, NNODE);
    fill_i32<<<blk(NNODE, T), T, 0, stream>>>(cand, 0x7FFFFFFF, NNODE);

    conv_weights<<<blk(3 * 4096, T), T, 0, stream>>>(lin1W, W2, lin2W, wt);

    xw_k<<<blk(NF, T), T, 0, stream>>>(x, W1, xw);
    deg_k<<<blk(NEDGE, T), T, 0, stream>>>(dst, deg);
    dinv_k<<<blk(NNODE, T), T, 0, stream>>>(deg, dinv);
    agg_k<<<blk((long)NEDGE * 64, T), T, 0, stream>>>(src, dst, dinv, xw, agg);
    gcn_fin<<<blk(NF, T), T, 0, stream>>>(agg, xw, dinv, b1, h, hb1, sq);

    build_adj<<<blk(NEDGE, T), T, 0, stream>>>(src, dst, adj, adjT);
    int nwaves = NG * MTILES * MTILES;
    pool_tiles<<<blk(nwaves, 8), 256, 0, stream>>>(adj, adjT, hb1, sq, cand);
    rep_k<<<blk(NNODE, T), T, 0, stream>>>(cand, rep);
    pool_accum<<<blk(NF, T), T, 0, stream>>>(rep, h, psum, pcount);
    pool_final<<<blk(NF, T), T, 0, stream>>>(psum, pcount, newx, hb2, valid);

    int gwaves = (NNODE / 16) * 4;
    gemm64<<<blk(gwaves, 8), 256, 0, stream>>>(hb2, wt, lin1b, h2, hb3, 1);

    gemm64<<<blk(gwaves, 8), 256, 0, stream>>>(hb3, wt + 4096, nullptr, hw2, nullptr, 0);
    deg2_k<<<blk(NEDGE, T), T, 0, stream>>>(rep, dst, deg2);
    dinv_k<<<blk(NNODE, T), T, 0, stream>>>(deg2, dinv2);
    agg2_k<<<blk((long)NEDGE * 64, T), T, 0, stream>>>(rep, src, dst, dinv2, hw2, agg2);
    gcn_fin<<<blk(NF, T), T, 0, stream>>>(agg2, hw2, dinv2, b2, h3, hb4, nullptr);

    gemm64<<<blk(gwaves, 8), 256, 0, stream>>>(hb4, wt + 2 * 4096, lin2b, h4, nullptr, 1);

    pool_cls<<<NG, 64, 0, stream>>>(h4, valid, clsW, clsb, out);
}
